// InteractionNetwork_1554778161262
// MI455X (gfx1250) — matrix operand order repaired and checked by reading, not compiled, not run on hardware
//
#include <hip/hip_runtime.h>
#include <hip/hip_bf16.h>

typedef __attribute__((ext_vector_type(16))) __bf16 v16bf;
typedef __attribute__((ext_vector_type(8)))  __bf16 v8bf;
typedef __attribute__((ext_vector_type(8)))  float  v8f;

#define D_FEAT 128
#define H_HID  512

static __device__ __forceinline__ v16bf cat16(v8bf lo, v8bf hi) {
  return __builtin_shufflevector(lo, hi, 0,1,2,3,4,5,6,7,8,9,10,11,12,13,14,15);
}
static __device__ __forceinline__ v8f bc8(float v) {
  return (v8f){v, v, v, v, v, v, v, v};
}
static __device__ __forceinline__ v8f wmma_bf16(v16bf a, v16bf b, v8f c) {
  // D = A(16x32 bf16) * B(32x16 bf16) + C(16x16 f32)
  return __builtin_amdgcn_wmma_f32_16x16x32_bf16(false, a, false, b, (short)0, c,
                                                 false, false);
}
static __device__ __forceinline__ v8bf cvt8(const float* __restrict__ p) {
  float4 f0 = *(const float4*)(p);
  float4 f1 = *(const float4*)(p + 4);
  return (v8bf){(__bf16)f0.x, (__bf16)f0.y, (__bf16)f0.z, (__bf16)f0.w,
                (__bf16)f1.x, (__bf16)f1.y, (__bf16)f1.z, (__bf16)f1.w};
}

// ---------------------------------------------------------------------------
// Prep kernels
// ---------------------------------------------------------------------------
__global__ void k_zero_f32(float* p, int n) {
  int i = blockIdx.x * blockDim.x + threadIdx.x;
  if (i < n) p[i] = 0.0f;
}

__global__ void k_cvt_bf16(const float* __restrict__ src, __bf16* __restrict__ dst, int n) {
  int i = blockIdx.x * blockDim.x + threadIdx.x;
  if (i < n) dst[i] = (__bf16)src[i];
}

// src is [K][N] row-major f32; dst is [N][K] row-major bf16 (N-major for B frags)
__global__ void k_transpose_bf16(const float* __restrict__ src, __bf16* __restrict__ dst,
                                 int K, int N) {
  int i = blockIdx.x * blockDim.x + threadIdx.x;
  if (i < K * N) {
    int n = i / K;
    int k = i - n * K;
    dst[(long)n * K + k] = (__bf16)src[(long)k * N + n];
  }
}

// ---------------------------------------------------------------------------
// Edge kernel: 64 edges (4 M-tiles) per workgroup, 8 wave32.
//   hidden = relu([x[s] | x[r] | e] @ We1 + be1)   (64x384 @ 384x512)
//   eout   = hidden @ We2 + be2                    (64x512 @ 512x128)
//   agg[recv] += eout  (f32 global atomics)
// Each B fragment (2 x global_load_b128) is reused across 4 M-tiles.
// ---------------------------------------------------------------------------
#define MT        4     // M-tiles of 16 per workgroup (64 rows)
#define A1_STRIDE 392   // 384 + 8 pad (keeps 16B alignment of 8-half chunks)
#define H_STRIDE  520   // 512 + 8 pad

__global__ __launch_bounds__(256) void edge_kernel(
    const __bf16* __restrict__ xb,
    const long long* __restrict__ senders,
    const long long* __restrict__ receivers,
    const float* __restrict__ edge_attr,
    const __bf16* __restrict__ Wt1,   // [512][384] bf16, N-major
    const float* __restrict__ be1,
    const __bf16* __restrict__ Wt2,   // [128][512] bf16, N-major
    const float* __restrict__ be2,
    float* __restrict__ out_edge,
    float* __restrict__ agg,
    int E) {
  __shared__ __align__(16) __bf16 Abuf[64 * A1_STRIDE];   // ~49 KB
  __shared__ __align__(16) __bf16 Hbuf[64 * H_STRIDE];    // ~65 KB
  __shared__ int sendS[64];
  __shared__ int recvS[64];

  const int tid   = threadIdx.x;
  const int wave  = tid >> 5;
  const int lane  = tid & 31;
  const int mrow  = lane & 15;
  const int hiK   = lane >> 4;          // 0 for lanes 0-15, 1 for lanes 16-31
  const int ebase = blockIdx.x * 64;

  if (tid < 64) {
    int eid = ebase + tid;
    if (eid >= E) eid = E - 1;
    recvS[tid] = (int)receivers[eid];
    sendS[tid] = (int)senders[eid];
  }
  __syncthreads();

  // Stage A tile [64][384] bf16, 8 columns (16B) per step:
  // cols 0-127 = x[sender], 128-255 = x[receiver], 256-383 = edge_attr (cvt).
  for (int gidx = tid; gidx < 64 * 48; gidx += 256) {
    int m = gidx / 48;
    int c = (gidx - m * 48) * 8;
    int eid = ebase + m;
    if (eid >= E) eid = E - 1;
    v8bf v;
    if (c < 128)       v = *(const v8bf*)(xb + (long)sendS[m] * D_FEAT + c);
    else if (c < 256)  v = *(const v8bf*)(xb + (long)recvS[m] * D_FEAT + (c - 128));
    else               v = cvt8(edge_attr + (long)eid * D_FEAT + (c - 256));
    *(v8bf*)(Abuf + m * A1_STRIDE + c) = v;
  }
  __syncthreads();

  // ---- GEMM1: [64x384] @ [384x512] -> relu -> Hbuf (bf16) ----
  {
    v8f acc[4][MT];
#pragma unroll
    for (int i = 0; i < 4; ++i) {
      const float bias = be1[(wave + i * 8) * 16 + mrow];
#pragma unroll
      for (int mt = 0; mt < MT; ++mt) acc[i][mt] = bc8(bias);
    }
    for (int kb = 0; kb < 384; kb += 32) {
      v16bf afrag[MT];
#pragma unroll
      for (int mt = 0; mt < MT; ++mt) {
        const __bf16* ap = Abuf + (mt * 16 + mrow) * A1_STRIDE + kb + hiK * 8;
        afrag[mt] = cat16(*(const v8bf*)ap, *(const v8bf*)(ap + 16));
      }
#pragma unroll
      for (int i = 0; i < 4; ++i) {
        const int ncol = (wave + i * 8) * 16 + mrow;
        const __bf16* bp = Wt1 + (long)ncol * 384 + kb + hiK * 8;
        v16bf bfrag = cat16(*(const v8bf*)bp, *(const v8bf*)(bp + 16));
#pragma unroll
        for (int mt = 0; mt < MT; ++mt)
          acc[i][mt] = wmma_bf16(afrag[mt], bfrag, acc[i][mt]);
      }
    }
#pragma unroll
    for (int i = 0; i < 4; ++i) {
      const int ncol = (wave + i * 8) * 16 + mrow;
#pragma unroll
      for (int mt = 0; mt < MT; ++mt)
#pragma unroll
        for (int r = 0; r < 8; ++r) {
          float v = acc[i][mt][r];
          v = v > 0.0f ? v : 0.0f;
          Hbuf[(mt * 16 + r + hiK * 8) * H_STRIDE + ncol] = (__bf16)v;
        }
    }
  }
  __syncthreads();

  // ---- GEMM2: [64x512] @ [512x128]; one 16-wide N tile per wave ----
  {
    const int ncol = wave * 16 + mrow;
    v8f acc[MT];
    const float bias = be2[ncol];
#pragma unroll
    for (int mt = 0; mt < MT; ++mt) acc[mt] = bc8(bias);
    for (int kb = 0; kb < 512; kb += 32) {
      const __bf16* bp = Wt2 + (long)ncol * 512 + kb + hiK * 8;
      v16bf bfrag = cat16(*(const v8bf*)bp, *(const v8bf*)(bp + 16));
#pragma unroll
      for (int mt = 0; mt < MT; ++mt) {
        const __bf16* ap = Hbuf + (mt * 16 + mrow) * H_STRIDE + kb + hiK * 8;
        v16bf afrag = cat16(*(const v8bf*)ap, *(const v8bf*)(ap + 16));
        acc[mt] = wmma_bf16(afrag, bfrag, acc[mt]);
      }
    }
#pragma unroll
    for (int mt = 0; mt < MT; ++mt)
#pragma unroll
      for (int r = 0; r < 8; ++r) {
        const int m   = mt * 16 + r + hiK * 8;
        const int eid = ebase + m;
        if (eid < E) {
          const float v = acc[mt][r];
          out_edge[(long)eid * D_FEAT + ncol] = v;
          atomicAdd(&agg[(long)recvS[m] * D_FEAT + ncol], v);
        }
      }
  }
}

// ---------------------------------------------------------------------------
// Node kernel: 64 nodes per workgroup.
//   out = relu([x | agg] @ Wn1 + bn1) @ Wn2 + bn2
// ---------------------------------------------------------------------------
#define A2_STRIDE 264   // 256 + 8 pad

__global__ __launch_bounds__(256) void node_kernel(
    const __bf16* __restrict__ xb,
    const float* __restrict__ agg,
    const __bf16* __restrict__ Wt3,   // [512][256] bf16, N-major
    const float* __restrict__ bn1,
    const __bf16* __restrict__ Wt4,   // [128][512] bf16, N-major
    const float* __restrict__ bn2,
    float* __restrict__ out_nodes,
    int Nn) {
  __shared__ __align__(16) __bf16 Abuf[64 * A2_STRIDE];   // ~33 KB
  __shared__ __align__(16) __bf16 Hbuf[64 * H_STRIDE];    // ~65 KB

  const int tid   = threadIdx.x;
  const int wave  = tid >> 5;
  const int lane  = tid & 31;
  const int mrow  = lane & 15;
  const int hiK   = lane >> 4;
  const int nbase = blockIdx.x * 64;

  // Stage A tile [64][256]: cols 0-127 = x, 128-255 = aggregated messages.
  for (int gidx = tid; gidx < 64 * 32; gidx += 256) {
    int m = gidx >> 5;
    int c = (gidx & 31) * 8;
    int node = nbase + m;
    if (node >= Nn) node = Nn - 1;
    v8bf v;
    if (c < 128) v = *(const v8bf*)(xb + (long)node * D_FEAT + c);
    else         v = cvt8(agg + (long)node * D_FEAT + (c - 128));
    *(v8bf*)(Abuf + m * A2_STRIDE + c) = v;
  }
  __syncthreads();

  // ---- GEMM1: [64x256] @ [256x512] -> relu -> Hbuf ----
  {
    v8f acc[4][MT];
#pragma unroll
    for (int i = 0; i < 4; ++i) {
      const float bias = bn1[(wave + i * 8) * 16 + mrow];
#pragma unroll
      for (int mt = 0; mt < MT; ++mt) acc[i][mt] = bc8(bias);
    }
    for (int kb = 0; kb < 256; kb += 32) {
      v16bf afrag[MT];
#pragma unroll
      for (int mt = 0; mt < MT; ++mt) {
        const __bf16* ap = Abuf + (mt * 16 + mrow) * A2_STRIDE + kb + hiK * 8;
        afrag[mt] = cat16(*(const v8bf*)ap, *(const v8bf*)(ap + 16));
      }
#pragma unroll
      for (int i = 0; i < 4; ++i) {
        const int ncol = (wave + i * 8) * 16 + mrow;
        const __bf16* bp = Wt3 + (long)ncol * 256 + kb + hiK * 8;
        v16bf bfrag = cat16(*(const v8bf*)bp, *(const v8bf*)(bp + 16));
#pragma unroll
        for (int mt = 0; mt < MT; ++mt)
          acc[i][mt] = wmma_bf16(afrag[mt], bfrag, acc[i][mt]);
      }
    }
#pragma unroll
    for (int i = 0; i < 4; ++i) {
      const int ncol = (wave + i * 8) * 16 + mrow;
#pragma unroll
      for (int mt = 0; mt < MT; ++mt)
#pragma unroll
        for (int r = 0; r < 8; ++r) {
          float v = acc[i][mt][r];
          v = v > 0.0f ? v : 0.0f;
          Hbuf[(mt * 16 + r + hiK * 8) * H_STRIDE + ncol] = (__bf16)v;
        }
    }
  }
  __syncthreads();

  // ---- GEMM2: [64x512] @ [512x128] ----
  {
    const int ncol = wave * 16 + mrow;
    v8f acc[MT];
    const float bias = bn2[ncol];
#pragma unroll
    for (int mt = 0; mt < MT; ++mt) acc[mt] = bc8(bias);
    for (int kb = 0; kb < 512; kb += 32) {
      const __bf16* bp = Wt4 + (long)ncol * 512 + kb + hiK * 8;
      v16bf bfrag = cat16(*(const v8bf*)bp, *(const v8bf*)(bp + 16));
#pragma unroll
      for (int mt = 0; mt < MT; ++mt) {
        const __bf16* ap = Hbuf + (mt * 16 + mrow) * H_STRIDE + kb + hiK * 8;
        v16bf afrag = cat16(*(const v8bf*)ap, *(const v8bf*)(ap + 16));
        acc[mt] = wmma_bf16(afrag, bfrag, acc[mt]);
      }
    }
#pragma unroll
    for (int mt = 0; mt < MT; ++mt)
#pragma unroll
      for (int r = 0; r < 8; ++r) {
        const int node = nbase + mt * 16 + r + hiK * 8;
        if (node < Nn) out_nodes[(long)node * D_FEAT + ncol] = acc[mt][r];
      }
  }
}

// ---------------------------------------------------------------------------
// Launch
// ---------------------------------------------------------------------------
extern "C" void kernel_launch(void* const* d_in, const int* in_sizes, int n_in,
                              void* d_out, int out_size, void* d_ws, size_t ws_size,
                              hipStream_t stream) {
  const float*     x   = (const float*)d_in[0];
  const long long* ei  = (const long long*)d_in[1];   // [2][E] int64
  const float*     ea  = (const float*)d_in[2];
  const float*     We1 = (const float*)d_in[3];
  const float*     be1 = (const float*)d_in[4];
  const float*     We2 = (const float*)d_in[5];
  const float*     be2 = (const float*)d_in[6];
  const float*     Wn1 = (const float*)d_in[7];
  const float*     bn1 = (const float*)d_in[8];
  const float*     Wn2 = (const float*)d_in[9];
  const float*     bn2 = (const float*)d_in[10];

  const int Nn = in_sizes[0] / D_FEAT;   // 10000
  const int E  = in_sizes[2] / D_FEAT;   // 320000

  const long long* senders   = ei;
  const long long* receivers = ei + E;

  // Output: [updated_nodes (Nn*128)] ++ [updated_edge_attr (E*128)]
  float* out_nodes = (float*)d_out;
  float* out_edges = (float*)d_out + (size_t)Nn * D_FEAT;

  // Workspace layout (256B-aligned sections)
  char*  ws   = (char*)d_ws;
  size_t off  = 0;
  float* agg  = (float*)(ws + off);  off += (size_t)Nn * D_FEAT * 4;      // 5.12 MB
  off = (off + 255) & ~(size_t)255;
  __bf16* Wt1 = (__bf16*)(ws + off); off += (size_t)512 * 384 * 2;        // We1^T
  off = (off + 255) & ~(size_t)255;
  __bf16* Wt2 = (__bf16*)(ws + off); off += (size_t)128 * 512 * 2;        // We2^T
  off = (off + 255) & ~(size_t)255;
  __bf16* Wt3 = (__bf16*)(ws + off); off += (size_t)512 * 256 * 2;        // Wn1^T
  off = (off + 255) & ~(size_t)255;
  __bf16* Wt4 = (__bf16*)(ws + off); off += (size_t)128 * 512 * 2;        // Wn2^T
  off = (off + 255) & ~(size_t)255;
  __bf16* xb  = (__bf16*)(ws + off); off += (size_t)Nn * D_FEAT * 2;      // x in bf16

  const int nAgg = Nn * D_FEAT;

  // Prep: zero aggregation buffer, convert x, transpose-convert weights to bf16.
  k_zero_f32<<<(nAgg + 255) / 256, 256, 0, stream>>>(agg, nAgg);
  k_cvt_bf16<<<(nAgg + 255) / 256, 256, 0, stream>>>(x, xb, nAgg);
  k_transpose_bf16<<<(384 * 512 + 255) / 256, 256, 0, stream>>>(We1, Wt1, 384, 512);
  k_transpose_bf16<<<(512 * 128 + 255) / 256, 256, 0, stream>>>(We2, Wt2, 512, 128);
  k_transpose_bf16<<<(256 * 512 + 255) / 256, 256, 0, stream>>>(Wn1, Wt3, 256, 512);
  k_transpose_bf16<<<(512 * 128 + 255) / 256, 256, 0, stream>>>(Wn2, Wt4, 512, 128);

  // Fused edge MLP + scatter-add.
  const int eTiles = (E + 63) / 64;
  edge_kernel<<<eTiles, 256, 0, stream>>>(xb, senders, receivers, ea,
                                          Wt1, be1, Wt2, be2, out_edges, agg, E);

  // Fused node MLP.
  const int nTiles = (Nn + 63) / 64;
  node_kernel<<<nTiles, 256, 0, stream>>>(xb, agg, Wt3, bn1, Wt4, bn2, out_nodes, Nn);
}
